// Attention_44238163149402
// MI455X (gfx1250) — matrix operand order repaired and checked by reading, not compiled, not run on hardware
//
#include <hip/hip_runtime.h>

// MI455X (gfx1250) wave32 WMMA attention.
// ~412 GFLOP total vs ~0.6 GB HBM traffic -> matrix-core bound; f16 WMMA
// (v_wmma_f32_16x16x32_f16, f32 accumulate) with flash attention.
// GEMM data movement uses GLOBAL_LOAD_ASYNC_TO_LDS (ASYNCcnt) + double
// buffering when the toolchain exposes the builtin.

#define B_   4
#define S_   2048
#define DIM_ 2048
#define NH_  16
#define HD_  128

typedef __attribute__((ext_vector_type(16))) _Float16     v16h;
typedef __attribute__((ext_vector_type(8)))  float        v8f;
typedef __attribute__((ext_vector_type(4)))  unsigned int v4u;
typedef __attribute__((ext_vector_type(4)))  int          v4i;
typedef __attribute__((address_space(3)))    v4i          lds_v4i;

#if defined(__has_builtin)
#if __has_builtin(__builtin_amdgcn_global_load_async_to_lds_b128) && \
    __has_builtin(__builtin_amdgcn_s_wait_asynccnt)
#define HAVE_ASYNC_LDS 1
#endif
#endif
#ifndef HAVE_ASYNC_LDS
#define HAVE_ASYNC_LDS 0
#endif

union Frag {           // 32 bytes = 8 VGPRs per lane = one 16x32 f16 operand
  v16h     h;
  v4u      q[2];
  _Float16 e[16];
};

__device__ __forceinline__ v8f wmma16(const Frag& a, const Frag& b, v8f c) {
  return __builtin_amdgcn_wmma_f32_16x16x32_f16(false, a.h, false, b.h,
                                                (short)0, c, false, false);
}

#if HAVE_ASYNC_LDS
// 16-byte async copy global -> LDS (per lane), tracked by ASYNCcnt.
__device__ __forceinline__ void async_cp16(const _Float16* g, _Float16* l) {
  __builtin_amdgcn_global_load_async_to_lds_b128((v4i*)g, (lds_v4i*)l, 0, 0);
}
#endif

// ---------------------------------------------------------------- convert
__global__ void cvt_kernel(const float* __restrict__ src,
                           _Float16* __restrict__ dst, int n) {
  int i = blockIdx.x * blockDim.x + threadIdx.x;
  int stp = blockDim.x * gridDim.x;
  for (; i < n; i += stp) dst[i] = (_Float16)src[i];
}

// ---------------------------------------------------------------- GEMM
// C[M,N] = A[M,K] * Bw[N,K]^T  (both row-major along K)
// mode 0: RoPE epilogue, store (b,h,s,d) f16   (Q and K projections)
// mode 2: store V transposed (b,h,d,s) f16
// mode 3: store fp32 row-major M x N           (final @ wo^T)
#define BM 128
#define BN 128
#define BK 32
#define LDA 40   // padded LDS row stride (halfs) to spread banks

__global__ __launch_bounds__(256) void gemm_f16(
    const _Float16* __restrict__ A, const _Float16* __restrict__ Bw,
    void* __restrict__ Out, const float* __restrict__ cosT,
    const float* __restrict__ sinT, int M, int N, int K, int mode)
{
#if HAVE_ASYNC_LDS
  __shared__ _Float16 As[2][BM * LDA];
  __shared__ _Float16 Bs[2][BM * LDA];
#else
  __shared__ _Float16 As[1][BM * LDA];
  __shared__ _Float16 Bs[1][BM * LDA];
#endif
  const int tid  = threadIdx.x;
  const int lane = tid & 31;
  const int w    = tid >> 5;           // 8 waves
  const int wm   = (w & 1) * 64;       // wave tile: 64 x 32
  const int wn   = (w >> 1) * 32;
  const int m0   = blockIdx.y * BM;
  const int n0   = blockIdx.x * BN;
  const int fm   = lane & 15;
  const int hh   = lane >> 4;

  v8f acc[4][2] = {};

  const int lrow = tid >> 1;           // each thread stages 16+16 halfs
  const int lk   = (tid & 1) * 16;
  const _Float16* aRow = A  + (size_t)(m0 + lrow) * K + lk;
  const _Float16* bRow = Bw + (size_t)(n0 + lrow) * K + lk;

#if HAVE_ASYNC_LDS
  // ---- async double-buffered pipeline: one barrier per K-step ----
  {
    _Float16* ad = &As[0][lrow * LDA + lk];
    _Float16* bd = &Bs[0][lrow * LDA + lk];
    async_cp16(aRow, ad);     async_cp16(aRow + 8, ad + 8);
    async_cp16(bRow, bd);     async_cp16(bRow + 8, bd + 8);
  }
  for (int k0 = 0; k0 < K; k0 += BK) {
    const int cur = (k0 >> 5) & 1;
    __builtin_amdgcn_s_wait_asynccnt(0);   // this wave's copies landed in LDS
    __syncthreads();                       // everyone's copies landed
    if (k0 + BK < K) {
      _Float16* ad = &As[cur ^ 1][lrow * LDA + lk];
      _Float16* bd = &Bs[cur ^ 1][lrow * LDA + lk];
      const _Float16* ag = aRow + k0 + BK;
      const _Float16* bg = bRow + k0 + BK;
      async_cp16(ag, ad);     async_cp16(ag + 8, ad + 8);
      async_cp16(bg, bd);     async_cp16(bg + 8, bd + 8);
    }
    const _Float16* as = As[cur];
    const _Float16* bs = Bs[cur];
    Frag bf[2];                         // B-frag: lane=col, k = {8h+e, 16+8h+e}
#pragma unroll
    for (int j = 0; j < 2; ++j) {
      const _Float16* bp = bs + (wn + j * 16 + fm) * LDA + hh * 8;
      bf[j].q[0] = *(const v4u*)(bp);
      bf[j].q[1] = *(const v4u*)(bp + 16);
    }
#pragma unroll
    for (int i = 0; i < 4; ++i) {       // A-frag: lane=row, k = {8h+e, 16+8h+e}
      Frag af;
      const _Float16* ap = as + (wm + i * 16 + fm) * LDA + hh * 8;
      af.q[0] = *(const v4u*)(ap);
      af.q[1] = *(const v4u*)(ap + 16);
#pragma unroll
      for (int j = 0; j < 2; ++j) acc[i][j] = wmma16(af, bf[j], acc[i][j]);
    }
  }
#else
  // ---- fallback: synchronous staging through VGPRs ----
  for (int k0 = 0; k0 < K; k0 += BK) {
    const v4u* ag = (const v4u*)(aRow + k0);
    const v4u* bg = (const v4u*)(bRow + k0);
    v4u a0 = ag[0], a1 = ag[1];
    v4u b0 = bg[0], b1 = bg[1];
    __syncthreads();
    *(v4u*)(&As[0][lrow * LDA + lk])     = a0;
    *(v4u*)(&As[0][lrow * LDA + lk + 8]) = a1;
    *(v4u*)(&Bs[0][lrow * LDA + lk])     = b0;
    *(v4u*)(&Bs[0][lrow * LDA + lk + 8]) = b1;
    __syncthreads();
    Frag bf[2];
#pragma unroll
    for (int j = 0; j < 2; ++j) {
      const _Float16* bp = &Bs[0][(wn + j * 16 + fm) * LDA + hh * 8];
      bf[j].q[0] = *(const v4u*)(bp);
      bf[j].q[1] = *(const v4u*)(bp + 16);
    }
#pragma unroll
    for (int i = 0; i < 4; ++i) {
      Frag af;
      const _Float16* ap = &As[0][(wm + i * 16 + fm) * LDA + hh * 8];
      af.q[0] = *(const v4u*)(ap);
      af.q[1] = *(const v4u*)(ap + 16);
#pragma unroll
      for (int j = 0; j < 2; ++j) acc[i][j] = wmma16(af, bf[j], acc[i][j]);
    }
  }
#endif

  // C layout: lane holds col n = base + l%16, rows m = base + r + 8*(l>>4)
  const int rowB = m0 + wm + hh * 8;
  const int colB = n0 + wn + fm;

  if (mode == 3) {
    float* Of = (float*)Out;
#pragma unroll
    for (int i = 0; i < 4; ++i)
#pragma unroll
      for (int j = 0; j < 2; ++j)
#pragma unroll
        for (int r = 0; r < 8; ++r)
          Of[(size_t)(rowB + i * 16 + r) * N + (colB + j * 16)] = acc[i][j][r];
  } else if (mode == 2) {
    _Float16* Oh = (_Float16*)Out;      // V^T: (b, h, d, s)
#pragma unroll
    for (int i = 0; i < 4; ++i)
#pragma unroll
      for (int j = 0; j < 2; ++j)
#pragma unroll
        for (int r = 0; r < 8; ++r) {
          int m = rowB + i * 16 + r, n = colB + j * 16;
          int bb = m >> 11, s = m & (S_ - 1);
          int hd = n >> 7,  d = n & (HD_ - 1);
          Oh[((size_t)(bb * NH_ + hd) * HD_ + d) * S_ + s] = (_Float16)acc[i][j][r];
        }
  } else {
    _Float16* Oh = (_Float16*)Out;      // RoPE; lanes l, l^1 hold cols n, n^1
#pragma unroll
    for (int i = 0; i < 4; ++i)
#pragma unroll
      for (int j = 0; j < 2; ++j)
#pragma unroll
        for (int r = 0; r < 8; ++r) {
          int m = rowB + i * 16 + r, n = colB + j * 16;
          float v = acc[i][j][r];
          float p = __shfl_xor(v, 1);
          int s = m & (S_ - 1), d = n & (HD_ - 1);
          float c  = cosT[s * (HD_ / 2) + (d >> 1)];
          float sn = sinT[s * (HD_ / 2) + (d >> 1)];
          float res = (n & 1) ? (v * c + p * sn) : (v * c - p * sn);
          int bb = m >> 11, hd = n >> 7;
          Oh[((size_t)(bb * NH_ + hd) * S_ + s) * HD_ + d] = (_Float16)res;
        }
  }
}

// ---------------------------------------------------------------- flash attn
// Computes S^T = K*Q^T so softmax stats are per-lane (lane owns one query
// column) and P^T in the WMMA C layout maps directly onto the B operand of
// O^T = V^T * P^T (one shfl_xor(16) per register, no LDS, no barriers).
__global__ __launch_bounds__(256) void flash_f16(
    const _Float16* __restrict__ Qg, const _Float16* __restrict__ Kg,
    const _Float16* __restrict__ Vtg, _Float16* __restrict__ Og)
{
  const int lane = threadIdx.x & 31;
  const int w    = threadIdx.x >> 5;
  const int b = blockIdx.z, h = blockIdx.y;
  const int q0 = blockIdx.x * 128 + w * 16;  // 16 queries per wave
  const int fm = lane & 15;
  const int hh = lane >> 4;

  const _Float16* Q  = Qg  + (size_t)(b * NH_ + h) * S_ * HD_;
  const _Float16* Kp = Kg  + (size_t)(b * NH_ + h) * S_ * HD_;
  const _Float16* Vt = Vtg + (size_t)(b * NH_ + h) * HD_ * S_;

  Frag qf[4];                           // Q as B operand, resident in regs
#pragma unroll
  for (int t = 0; t < 4; ++t) {
    const _Float16* qp = Q + (size_t)(q0 + fm) * HD_ + t * 32 + hh * 8;
    qf[t].q[0] = *(const v4u*)qp;
    qf[t].q[1] = *(const v4u*)(qp + 16);
  }

  v8f ot[8] = {};                       // O^T: 128 d-rows x 16 q-cols
  float mx = -1e30f, sum = 0.f;
  const float scale = 0.08838834764831845f;  // 1/sqrt(128)
  const int qcol = q0 + fm;

  for (int kv0 = 0; kv0 < q0 + 16; kv0 += 64) {
    v8f st[4] = {};                     // S^T: 64 kv-rows x 16 q-cols
#pragma unroll
    for (int i = 0; i < 4; ++i) {
#pragma unroll
      for (int t = 0; t < 4; ++t) {
        Frag kf;                        // K as A operand
        const _Float16* kp = Kp + (size_t)(kv0 + i * 16 + fm) * HD_ + t * 32 + hh * 8;
        kf.q[0] = *(const v4u*)kp;
        kf.q[1] = *(const v4u*)(kp + 16);
        st[i] = wmma16(kf, qf[t], st[i]);
      }
    }
    // scale + causal mask + online softmax (per-column = per-lane stats)
    float tmax = -1e30f;
#pragma unroll
    for (int i = 0; i < 4; ++i)
#pragma unroll
      for (int r = 0; r < 8; ++r) {
        int kva = kv0 + i * 16 + hh * 8 + r;
        float v = st[i][r] * scale;
        v = (kva <= qcol) ? v : -1e30f;
        st[i][r] = v;
        tmax = fmaxf(tmax, v);
      }
    tmax = fmaxf(tmax, __shfl_xor(tmax, 16));
    float nmx  = fmaxf(mx, tmax);
    float corr = __expf(mx - nmx);
    mx = nmx;
    float psum = 0.f;
#pragma unroll
    for (int i = 0; i < 4; ++i)
#pragma unroll
      for (int r = 0; r < 8; ++r) {
        float p = __expf(st[i][r] - mx);
        st[i][r] = p;
        psum += p;
      }
    psum += __shfl_xor(psum, 16);
    sum = sum * corr + psum;
#pragma unroll
    for (int t = 0; t < 8; ++t)
#pragma unroll
      for (int r = 0; r < 8; ++r) ot[t][r] *= corr;

#pragma unroll
    for (int s2 = 0; s2 < 2; ++s2) {    // two 32-kv steps per block
      Frag pf;                          // P^T as B operand, built in-regs
#pragma unroll
      for (int r = 0; r < 8; ++r) {
        float a0 = st[2 * s2][r], a1 = st[2 * s2 + 1][r];
        float sendv = hh ? a0 : a1;
        float recv  = __shfl_xor(sendv, 16);
        float ownv  = hh ? a1 : a0;
        pf.e[r]     = (_Float16)(hh ? recv : ownv);
        pf.e[8 + r] = (_Float16)(hh ? ownv : recv);
      }
#pragma unroll
      for (int t = 0; t < 8; ++t) {
        Frag vf;                        // V^T as A operand (row-contiguous)
        const _Float16* vp = Vt + (size_t)(t * 16 + fm) * S_ + kv0 + s2 * 32 + hh * 16;
        vf.q[0] = *(const v4u*)vp;
        vf.q[1] = *(const v4u*)(vp + 8);
        ot[t] = wmma16(vf, pf, ot[t]);
      }
    }
  }

  float inv = 1.f / sum;
#pragma unroll
  for (int t = 0; t < 8; ++t)
#pragma unroll
    for (int r = 0; r < 8; ++r) {
      int d = t * 16 + hh * 8 + r;
      Og[((size_t)(b * S_ + qcol) * NH_ + h) * HD_ + d] = (_Float16)(ot[t][r] * inv);
    }
}

// ---------------------------------------------------------------- launch
extern "C" void kernel_launch(void* const* d_in, const int* in_sizes, int n_in,
                              void* d_out, int out_size, void* d_ws, size_t ws_size,
                              hipStream_t stream)
{
  const float* x  = (const float*)d_in[0];
  const float* wq = (const float*)d_in[1];
  const float* wk = (const float*)d_in[2];
  const float* wv = (const float*)d_in[3];
  const float* wo = (const float*)d_in[4];
  const float* fc = (const float*)d_in[5];
  const float* fs = (const float*)d_in[6];
  float* out = (float*)d_out;

  const size_t NX = (size_t)B_ * S_ * DIM_;   // 16.7M halfs
  const size_t NW = (size_t)DIM_ * DIM_;      // 4.2M halfs
  _Float16* xh  = (_Float16*)d_ws;
  _Float16* wqh = xh  + NX;
  _Float16* wkh = wqh + NW;
  _Float16* wvh = wkh + NW;
  _Float16* woh = wvh + NW;
  _Float16* qh  = woh + NW;
  _Float16* kh  = qh  + NX;
  _Float16* vth = kh  + NX;
  _Float16* oh  = xh;   // x no longer needed after QKV projections

  cvt_kernel<<<2048, 256, 0, stream>>>(x,  xh,  (int)NX);
  cvt_kernel<<<512,  256, 0, stream>>>(wq, wqh, (int)NW);
  cvt_kernel<<<512,  256, 0, stream>>>(wk, wkh, (int)NW);
  cvt_kernel<<<512,  256, 0, stream>>>(wv, wvh, (int)NW);
  cvt_kernel<<<512,  256, 0, stream>>>(wo, woh, (int)NW);

  dim3 g(DIM_ / BN, (B_ * S_) / BM);          // 16 x 64 blocks
  gemm_f16<<<g, 256, 0, stream>>>(xh, wqh, (void*)qh,  fc, fs, B_ * S_, DIM_, DIM_, 0);
  gemm_f16<<<g, 256, 0, stream>>>(xh, wkh, (void*)kh,  fc, fs, B_ * S_, DIM_, DIM_, 0);
  gemm_f16<<<g, 256, 0, stream>>>(xh, wvh, (void*)vth, fc, fs, B_ * S_, DIM_, DIM_, 2);

  flash_f16<<<dim3(S_ / 128, NH_, B_), 256, 0, stream>>>(qh, kh, vth, oh);

  gemm_f16<<<g, 256, 0, stream>>>(oh, woh, (void*)out, fc, fs, B_ * S_, DIM_, DIM_, 3);
}
